// RNNClassifier_74062416053142
// MI455X (gfx1250) — matrix operand order repaired and checked by reading, not compiled, not run on hardware
//
#include <hip/hip_runtime.h>
#include <cmath>

typedef __attribute__((ext_vector_type(16))) _Float16 v16h;
typedef __attribute__((ext_vector_type(8)))  _Float16 v8h;
typedef __attribute__((ext_vector_type(8)))  float    v8f;

#define HDIM 32
#define NCLS 10
#define SWZ_SWAPX16 0x401f   // group-of-32: xor_mask=0x10, and_mask=0x1f

#if __has_builtin(__builtin_amdgcn_tanhf)
#define TANH(v) __builtin_amdgcn_tanhf(v)
#else
#define TANH(v) tanhf(v)
#endif

__device__ __forceinline__ v8f wmma_f16(v16h a, v16h b, v8f c) {
    // D = A(16x32 f16) x B(32x16 f16) + C(16x16 f32)
    return __builtin_amdgcn_wmma_f32_16x16x32_f16(
        false, a, false, b, (short)0, c, false, false);
}

union HPack { v8h h; int i[4]; };
union BPack { v16h h; int i[8]; };

// Convert two post-tanh D tiles (f32) of S_new^T into the 32x16 f16 B operand
// for the next WMMA, using only a lane-pair swap (ln <-> ln+16). No LDS memory.
__device__ __forceinline__ v16h d_to_b(const v8f& t0, const v8f& t1, int hi) {
    HPack u0, u1, q0, q1; BPack b;
#pragma unroll
    for (int r = 0; r < 8; ++r) { u0.h[r] = (_Float16)t0[r]; u1.h[r] = (_Float16)t1[r]; }
#pragma unroll
    for (int j = 0; j < 4; ++j) {
        q0.i[j] = __builtin_amdgcn_ds_swizzle(u0.i[j], SWZ_SWAPX16);
        q1.i[j] = __builtin_amdgcn_ds_swizzle(u1.i[j], SWZ_SWAPX16);
    }
#pragma unroll
    for (int j = 0; j < 4; ++j) {
        b.i[j]     = hi ? q1.i[j] : u0.i[j];   // K elems 0..7
        b.i[4 + j] = hi ? u1.i[j] : q0.i[j];   // K elems 8..15
    }
    return b.h;
}

__global__ __launch_bounds__(32) void rnn2_wmma_kernel(
    const float* __restrict__ x,
    const float* __restrict__ Wih0, const float* __restrict__ Whh0,
    const float* __restrict__ bih0, const float* __restrict__ bhh0,
    const float* __restrict__ Wih1, const float* __restrict__ Whh1,
    const float* __restrict__ bih1, const float* __restrict__ bhh1,
    const float* __restrict__ fcw,  const float* __restrict__ fcb,
    float* __restrict__ out, int T)
{
    __shared__ float h2f[16 * HDIM];    // final layer-2 hidden for FC: [batch][feature]

    const int lane = threadIdx.x & 31;
    const int ln   = lane & 15;
    const int hi   = lane >> 4;
    const int m0   = blockIdx.x * 16;   // batch tile base (batch = WMMA N dim)

    // ---- Weights as A operands (static): A tile mt holds W rows mt*16+ln.
    //      A-layout element i <-> K = hi*16 + i.
    v16h wih0a[2], whh0a[2], wih1a[2], whh1a[2];
#pragma unroll
    for (int mt = 0; mt < 2; ++mt) {
        const int row = mt * 16 + ln;
#pragma unroll
        for (int i = 0; i < 16; ++i) {
            const int k = hi * 16 + i;
            wih0a[mt][i] = (_Float16)Wih0[row * HDIM + k];
            whh0a[mt][i] = (_Float16)Whh0[row * HDIM + k];
            wih1a[mt][i] = (_Float16)Wih1[row * HDIM + k];
            whh1a[mt][i] = (_Float16)Whh1[row * HDIM + k];
        }
    }
    // ---- Biases per D element: tile mt, element r <-> feature m = mt*16 + hi*8 + r.
    v8f bias1t[2], bias2t[2];
#pragma unroll
    for (int mt = 0; mt < 2; ++mt)
#pragma unroll
        for (int r = 0; r < 8; ++r) {
            const int m = mt * 16 + hi * 8 + r;
            bias1t[mt][r] = bih0[m] + bhh0[m];
            bias2t[mt][r] = bih1[m] + bhh1[m];
        }

    // ---- State S = h^T in B-format f16 registers; h0 = 0.
    v16h s1b = {}, s2b = {};
    v8f last20 = {}, last21 = {};

    const float* xrow = x + (size_t)(m0 + ln) * (size_t)T * HDIM + hi * 16;

    for (int t = 0; t < T; ++t) {
        const float* xp = xrow + (size_t)t * HDIM;
        __builtin_prefetch(xp + 8 * HDIM, 0, 1);   // stream x ~8 steps ahead

        // X_t^T in B-format: lane (ln,hi) holds features K = hi*16 + i of batch col ln
        // -> 16 contiguous floats, converted to f16.
        BPack xb;
#pragma unroll
        for (int i = 0; i < 16; ++i) xb.h[i] = (_Float16)xp[i];

        // ---- layer 1: S1 = tanh(Wih0 @ X^T + Whh0 @ S1 + b1)
        v8f c10 = bias1t[0], c11 = bias1t[1];
        c10 = wmma_f16(wih0a[0], xb.h, c10);
        c11 = wmma_f16(wih0a[1], xb.h, c11);
        c10 = wmma_f16(whh0a[0], s1b, c10);
        c11 = wmma_f16(whh0a[1], s1b, c11);
#pragma unroll
        for (int r = 0; r < 8; ++r) { c10[r] = TANH(c10[r]); c11[r] = TANH(c11[r]); }
        s1b = d_to_b(c10, c11, hi);

        // ---- layer 2: S2 = tanh(Wih1 @ S1 + Whh1 @ S2 + b2)
        v8f c20 = bias2t[0], c21 = bias2t[1];
        c20 = wmma_f16(wih1a[0], s1b, c20);
        c21 = wmma_f16(wih1a[1], s1b, c21);
        c20 = wmma_f16(whh1a[0], s2b, c20);
        c21 = wmma_f16(whh1a[1], s2b, c21);
#pragma unroll
        for (int r = 0; r < 8; ++r) { c20[r] = TANH(c20[r]); c21[r] = TANH(c21[r]); }
        s2b = d_to_b(c20, c21, hi);

        last20 = c20; last21 = c21;
    }

    // ---- final FC from f32 last h2: D tile mt, elem r -> feature mt*16+hi*8+r, batch ln.
#pragma unroll
    for (int r = 0; r < 8; ++r) {
        h2f[ln * HDIM + hi * 8 + r]      = last20[r];
        h2f[ln * HDIM + 16 + hi * 8 + r] = last21[r];
    }
    __syncthreads();

    for (int o = lane; o < 16 * NCLS; o += 32) {
        const int row = o / NCLS;
        const int cls = o - row * NCLS;
        float acc = fcb[cls];
#pragma unroll
        for (int k = 0; k < HDIM; ++k)
            acc += h2f[row * HDIM + k] * fcw[cls * HDIM + k];
        out[(m0 + row) * NCLS + cls] = acc;
    }
}

extern "C" void kernel_launch(void* const* d_in, const int* in_sizes, int n_in,
                              void* d_out, int out_size, void* d_ws, size_t ws_size,
                              hipStream_t stream) {
    (void)n_in; (void)out_size; (void)d_ws; (void)ws_size;
    const float* x    = (const float*)d_in[0];
    const float* Wih0 = (const float*)d_in[1];
    const float* Whh0 = (const float*)d_in[2];
    const float* bih0 = (const float*)d_in[3];
    const float* bhh0 = (const float*)d_in[4];
    const float* Wih1 = (const float*)d_in[5];
    const float* Whh1 = (const float*)d_in[6];
    const float* bih1 = (const float*)d_in[7];
    const float* bhh1 = (const float*)d_in[8];
    const float* fcw  = (const float*)d_in[9];
    const float* fcb  = (const float*)d_in[10];
    float* out = (float*)d_out;

    const int T = 2048;
    const int B = in_sizes[0] / (T * HDIM);   // 512

    dim3 grid(B / 16), block(32);             // one wave per 16 batch rows
    hipLaunchKernelGGL(rnn2_wmma_kernel, grid, block, 0, stream,
                       x, Wih0, Whh0, bih0, bhh0,
                       Wih1, Whh1, bih1, bhh1, fcw, fcb, out, T);
}
